// ScaledDotProductAttention_55929064128822
// MI455X (gfx1250) — matrix operand order repaired and checked by reading, not compiled, not run on hardware
//
#include <hip/hip_runtime.h>
#include <hip/hip_bf16.h>

// ---- problem constants (fixed by reference) ----
#define BATCH 64
#define LSEQ  1024
#define DK    64
#define PPOS  (2 * LSEQ - 1)        // 2047
#define INV_TEMPER 0.125f           // 1/sqrt(64)

typedef __attribute__((ext_vector_type(16))) _Float16 v16h;
typedef __attribute__((ext_vector_type(8)))  _Float16 v8h;
typedef __attribute__((ext_vector_type(8)))  float    v8f;
typedef __attribute__((ext_vector_type(4)))  float    v4f;

#define W_STRIDE     1040           // floats per W row (need 1039)
#define P_STRIDE     1040           // halves per probs row (aliases W buffer)
#define S_STRIDE     1024           // floats per score row

// WMMA wrapper: D = A(16x32 f16) * B(32x16 f16) + C(f32)
__device__ __forceinline__ v8f wmma16x16x32(v16h a, v16h b, v8f c) {
  return __builtin_amdgcn_wmma_f32_16x16x32_f16(
      /*neg_a=*/false, a, /*neg_b=*/false, b,
      /*c_mod=*/(short)0, c, /*reuse_a=*/false, /*reuse_b=*/false);
}

// A fragment (16x32 f16) from an LDS row of >=32 f16 starting at kbase.
// Contiguous layout: lane-half h, element e -> K=kbase+16h+e. Two 16B ds loads.
__device__ __forceinline__ v16h load_a_lds(const _Float16* rowbase, int kbase, int h) {
  const v8h* p0 = (const v8h*)(rowbase + kbase + 16 * h);
  const v8h* p1 = (const v8h*)(rowbase + kbase + 16 * h + 8);
  v8h lo = *p0;
  v8h hi = *p1;
  v16h a;
#pragma unroll
  for (int e = 0; e < 8; ++e) { a[e] = lo[e]; a[8 + e] = hi[e]; }
  return a;
}

// B fragment (32x16 f16) from 16 contiguous global f32 (K = 16h .. 16h+15
// for this lane-half, column n = lane&15). 64B-aligned: four b128 loads.
__device__ __forceinline__ v16h load_b_f32x16(const float* src) {
  const v4f* s4 = (const v4f*)src;
  v16h b;
#pragma unroll
  for (int q4 = 0; q4 < 4; ++q4) {
    v4f x = s4[q4];
#pragma unroll
    for (int e = 0; e < 4; ++e) b[4 * q4 + e] = (_Float16)x[e];
  }
  return b;
}

__global__ void __launch_bounds__(256)
dpa_attn_kernel(const float* __restrict__ q,
                const float* __restrict__ kmat,
                const float* __restrict__ vmat,
                const float* __restrict__ pos,
                const unsigned char* __restrict__ mask,   // jax bool = 1 byte
                float* __restrict__ out) {
  // LDS: q tile (f16), reversed-q tile (f16), scores/exp (f32), W/probs alias
  __shared__ _Float16 lds_q [16 * DK];
  __shared__ _Float16 lds_qr[16 * DK];
  __shared__ float    lds_s [16 * S_STRIDE];
  __shared__ float    lds_w [16 * W_STRIDE];

  const int i0   = blockIdx.x * 16;      // row tile
  const int b    = blockIdx.y;           // batch
  const int tid  = threadIdx.x;
  const int lane = tid & 31;
  const int wave = tid >> 5;             // 8 waves
  const int nsub = lane & 15;            // column n of B/C/D fragments
  const int h    = lane >> 4;            // lane-half

  const size_t qBase = (size_t)b * LSEQ * DK;
  const size_t pBase = (size_t)b * PPOS * DK;
  const size_t mBase = (size_t)b * LSEQ * LSEQ;
  const size_t attnBase = (size_t)BATCH * LSEQ * DK;  // attn follows output in d_out

  // ---- phase 0: stage Q tile and reversed-Q tile as f16 ----
  for (int idx = tid; idx < 16 * DK; idx += 256) {
    const int s = idx >> 6, d = idx & 63;
    lds_q [idx] = (_Float16)q[qBase + (size_t)(i0 + s) * DK + d];
    lds_qr[idx] = (_Float16)q[qBase + (size_t)(LSEQ - 1 - (i0 + s)) * DK + d];
  }
  __syncthreads();

  // A fragments (held across phases): row m = lane&15 of the tile
  v16h aq[2], aqr[2];
#pragma unroll
  for (int kb = 0; kb < 2; ++kb) {
    aq [kb] = load_a_lds(lds_q  + (lane & 15) * DK, 32 * kb, h);
    aqr[kb] = load_a_lds(lds_qr + (lane & 15) * DK, 32 * kb, h);
  }

  // ---- phase 1: W[s, m-i0] = qrev . p[m], m in [i0, i0+1040) ----
  for (int tw = wave; tw < 65; tw += 8) {
    const int m0 = i0 + tw * 16;
    int prow = m0 + nsub;
    if (prow > PPOS - 1) prow = PPOS - 1;   // one-row overread clamp (value unused)
    v8f acc = {};
#pragma unroll
    for (int kb = 0; kb < 2; ++kb) {
      v16h bp = load_b_f32x16(pos + pBase + (size_t)prow * DK + 32 * kb + 16 * h);
      acc = wmma16x16x32(aqr[kb], bp, acc);
    }
#pragma unroll
    for (int vv = 0; vv < 8; ++vv)
      lds_w[(vv + 8 * h) * W_STRIDE + tw * 16 + nsub] = acc[vv];
  }
  __syncthreads();

  // ---- phase 2: scores = (Q.K^T + W-shift) / 8, mask -> -inf ----
  for (int jt = wave; jt < 64; jt += 8) {
    const int j0 = jt * 16;
    v8f acc = {};
#pragma unroll
    for (int kb = 0; kb < 2; ++kb) {
      v16h bk = load_b_f32x16(kmat + qBase + (size_t)(j0 + nsub) * DK + 32 * kb + 16 * h);
      acc = wmma16x16x32(aq[kb], bk, acc);
    }
    const int j = j0 + nsub;
#pragma unroll
    for (int vv = 0; vv < 8; ++vv) {
      const int s = vv + 8 * h;
      float sc = (acc[vv] + lds_w[s * W_STRIDE + s + j]) * INV_TEMPER;
      // mask is streamed once: non-temporal load (TH=NT)
      const unsigned char mbit =
          __builtin_nontemporal_load(mask + mBase + (size_t)(i0 + s) * LSEQ + j);
      if (mbit) sc = -__builtin_inff();
      lds_s[s * S_STRIDE + j] = sc;
    }
  }
  __syncthreads();

  // ---- phase 3: softmax (2 rows per wave); write attn + f16 probs ----
  // exp values are cached back into lds_s so v_exp runs once per element.
  _Float16* lds_p = (_Float16*)lds_w;     // W dead; alias as probs (16 x P_STRIDE f16)
#pragma unroll
  for (int rr = 0; rr < 2; ++rr) {
    const int r = 2 * wave + rr;
    float* row = lds_s + r * S_STRIDE;
    float mx = -__builtin_inff();
    for (int c = lane; c < LSEQ; c += 32) mx = fmaxf(mx, row[c]);
#pragma unroll
    for (int off = 16; off > 0; off >>= 1) mx = fmaxf(mx, __shfl_xor(mx, off, 32));
    float ssum = 0.f;
    for (int c = lane; c < LSEQ; c += 32) {
      const float e = __expf(row[c] - mx);
      ssum += e;
      row[c] = e;                         // cache exp for the write pass
    }
#pragma unroll
    for (int off = 16; off > 0; off >>= 1) ssum += __shfl_xor(ssum, off, 32);
    const float inv = 1.0f / ssum;
    float* attn_row = out + attnBase + mBase + (size_t)(i0 + r) * LSEQ;
    for (int c = lane; c < LSEQ; c += 32) {
      const float p = row[c] * inv;
      __builtin_nontemporal_store(p, attn_row + c);   // streamed output (TH=NT)
      lds_p[r * P_STRIDE + c] = (_Float16)p;
    }
  }
  __syncthreads();

  // ---- phase 4: out = probs . V  — all 8 waves ----
  // wave -> (dcol tile = wave&3, m-half = wave>>2); waves 4..7 deposit partial
  // f32 tiles into lds_s (dead after phase 3), waves 0..3 reduce + store.
  {
    const int dtile = wave & 3;
    const int half  = wave >> 2;
    const int dcol  = dtile * 16 + nsub;
    v8f acc = {};
    for (int c = 0; c < 16; ++c) {
      const int m0 = (half * 16 + c) * 32;
      v16h ap = load_a_lds(lds_p + (lane & 15) * P_STRIDE, m0, h);
      v16h bv;
#pragma unroll
      for (int e = 0; e < 16; ++e)
        bv[e] = (_Float16)vmat[qBase + (size_t)(m0 + 16 * h + e) * DK + dcol];
      acc = wmma16x16x32(ap, bv, acc);
    }
    float* lds_part = lds_s;              // 16 rows x 64 cols partial buffer
    if (half == 1) {
#pragma unroll
      for (int vv = 0; vv < 8; ++vv)
        lds_part[(vv + 8 * h) * DK + dcol] = acc[vv];
    }
    __syncthreads();
    if (half == 0) {
#pragma unroll
      for (int vv = 0; vv < 8; ++vv) {
        const float r = acc[vv] + lds_part[(vv + 8 * h) * DK + dcol];
        __builtin_nontemporal_store(
            r, out + qBase + (size_t)(i0 + vv + 8 * h) * DK + dcol);
      }
    }
  }
}

extern "C" void kernel_launch(void* const* d_in, const int* in_sizes, int n_in,
                              void* d_out, int out_size, void* d_ws, size_t ws_size,
                              hipStream_t stream) {
  const float* q   = (const float*)d_in[0];
  const float* k   = (const float*)d_in[1];
  const float* v   = (const float*)d_in[2];
  const float* pos = (const float*)d_in[3];
  const unsigned char* mask = (const unsigned char*)d_in[4];
  float* out = (float*)d_out;

  dim3 grid(LSEQ / 16, BATCH);   // 64 x 64 workgroups
  dim3 block(256);               // 8 wave32s
  dpa_attn_kernel<<<grid, block, 0, stream>>>(q, k, v, pos, mask, out);
}
